// ECGSpikingEncoder_14336600834539
// MI455X (gfx1250) — matrix operand order repaired and checked by reading, not compiled, not run on hardware
//
#include <hip/hip_runtime.h>
#include <hip/hip_bf16.h>

typedef __bf16 bf16_t;
typedef __attribute__((ext_vector_type(16))) __bf16 v16bf;
typedef __attribute__((ext_vector_type(8)))  __bf16 v8bf;
typedef __attribute__((ext_vector_type(8)))  float  v8f;

// Problem constants (from reference setup_inputs)
#define NB   512      // batch
#define NC   12       // in channels
#define NL   5000     // length
#define NE   128      // conv out / embed
#define NH1  128
#define NH2  128
#define NP   50       // kernel = stride
#define NT   100      // (5000-50)/50 + 1
#define K1   600      // NC*NP
#define K1P  608      // padded to multiple of 32
#define MROWS (NT*NB) // 51200 GEMM rows, row index = t*NB + b

// -------- workspace layout (bytes, all 256B aligned) --------
#define OFF_APK   ((size_t)0)                              // bf16 [MROWS][K1P]
#define OFF_H12   ((size_t)62259200)                       // f32  [NT][NB][128]
#define OFF_S1    ((size_t)(62259200 + 26214400))          // bf16 [NT][NB][NH1]
#define OFF_W1BF  ((size_t)(62259200 + 26214400 + 13107200)) // bf16 [NH1][K1P]
#define OFF_FC2BF (OFF_W1BF + 155648)                      // bf16 [NH2][NH1]
#define OFF_B1EFF (OFF_FC2BF + 32768)                      // f32  [NH1]

// ---- 1. Fold conv into fc1: W1eff = fc1_w @ conv_w, b1eff = fc1_b + fc1_w @ conv_b,
//         plus fc2_w -> bf16. Tiny (20 MFLOP total).
__global__ void prep_weights(const float* __restrict__ conv_w, const float* __restrict__ conv_b,
                             const float* __restrict__ fc1_w, const float* __restrict__ fc1_b,
                             const float* __restrict__ fc2_w,
                             bf16_t* __restrict__ W1bf, float* __restrict__ b1eff,
                             bf16_t* __restrict__ fc2bf) {
    int idx = blockIdx.x * blockDim.x + threadIdx.x;
    const int NW1 = NH1 * K1P;
    if (idx < NW1) {
        int h = idx / K1P, k = idx % K1P;
        float s = 0.0f;
        if (k < K1) {
            for (int e = 0; e < NE; ++e)
                s = fmaf(fc1_w[h * NE + e], conv_w[e * K1 + k], s);
        }
        W1bf[idx] = (bf16_t)s;
    } else if (idx < NW1 + NH1) {
        int h = idx - NW1;
        float s = fc1_b[h];
        for (int e = 0; e < NE; ++e)
            s = fmaf(fc1_w[h * NE + e], conv_b[e], s);
        b1eff[h] = s;
    } else if (idx < NW1 + NH1 + NH2 * NH1) {
        int i = idx - NW1 - NH1;
        fc2bf[i] = (bf16_t)fc2_w[i];
    }
}

// ---- 2. Pack x windows into bf16 A-matrix rows: Apk[t*NB+b][c*50+p] = x[b][c][t*50+p]
__global__ void pack_A(const float* __restrict__ x, bf16_t* __restrict__ Apk) {
    size_t idx = (size_t)blockIdx.x * blockDim.x + threadIdx.x;
    if (idx >= (size_t)MROWS * K1P) return;
    int k   = (int)(idx % K1P);
    int row = (int)(idx / K1P);
    int t = row / NB;
    int b = row % NB;
    float v = 0.0f;
    if (k < K1) {
        int c = k / NP;
        int p = k % NP;
        v = x[(size_t)b * (NC * NL) + (size_t)c * NL + t * NP + p];
    }
    Apk[idx] = (bf16_t)v;
}

// ---- 3. WMMA GEMM: D[m][n] = sum_k A[m][k]*Bt[n][k] + bias[n]
// A : bf16 [M][KPAD] row-major. Bt: bf16 [128][KPAD] (weights, row = output col).
// Block = 256 threads = 8 waves; each wave owns one 16-row M-tile and
// register-tiles all 8 N-tiles (v8f acc[8]). The full B panel is DMA'd into
// LDS once per block with async global->LDS (ASYNCcnt), then read via ds_load.
template<int KPAD>
__global__ __launch_bounds__(256)
void gemm_wmma_bf16(const bf16_t* __restrict__ A, const bf16_t* __restrict__ Bt,
                    const float* __restrict__ bias, float* __restrict__ D) {
    __shared__ bf16_t bsh[128 * KPAD];   // 155,648 B (K1P) / 32 KB (128) < 320 KB

    const int tid  = threadIdx.x;
    const int lane = tid & 31;
    const int wave = tid >> 5;
    const int half = lane >> 4;   // which 16-lane half of the wave
    const int l    = lane & 15;

    // ---- stage B panel into LDS via CDNA5 async DMA (tracked by ASYNCcnt) ----
    {
        const int nbytes = 128 * KPAD * 2;   // multiple of 256*16 for both KPADs
        unsigned lbase = (unsigned)(size_t)&bsh[0];
        for (int off = tid * 16; off < nbytes; off += 256 * 16) {
            unsigned laddr = lbase + (unsigned)off;
            const char* g = (const char*)Bt + off;
            asm volatile("global_load_async_to_lds_b128 %0, %1, off"
                         :: "v"(laddr), "v"(g) : "memory");
        }
        asm volatile("s_wait_asynccnt 0x0" ::: "memory");
        __syncthreads();
    }

    const int m0 = (blockIdx.x * 8 + wave) * 16;
    // A-matrix lane view (ISA 7.12.2, 16-bit A 16x32): row M = l;
    // lanes 0-15 hold K = {0..7, 16..23}, lanes 16-31 hold K = {8..15, 24..31}
    const bf16_t* ap = A + (size_t)(m0 + l) * KPAD + half * 8;

    v8f acc[8] = {};
    for (int k0 = 0; k0 < KPAD; k0 += 32) {
        __builtin_prefetch(ap + k0 + 64, 0, 1);   // global_prefetch_b8
        v8bf a_lo = *(const v8bf*)(ap + k0);
        v8bf a_hi = *(const v8bf*)(ap + k0 + 16);
        v16bf a;
        #pragma unroll
        for (int i = 0; i < 8; ++i) { a[i] = a_lo[i]; a[i + 8] = a_hi[i]; }
        #pragma unroll
        for (int nt = 0; nt < 8; ++nt) {
            // B-matrix lane view (32x16, col per lane): col N = l;
            // lanes 0-15 hold K = {0..7, 16..23}, lanes 16-31 hold K = {8..15, 24..31}
            const bf16_t* bp = bsh + (size_t)(nt * 16 + l) * KPAD + half * 8 + k0;
            v8bf b_lo = *(const v8bf*)(bp);       // ds_load_b128
            v8bf b_hi = *(const v8bf*)(bp + 16);  // ds_load_b128
            v16bf b;
            #pragma unroll
            for (int i = 0; i < 8; ++i) { b[i] = b_lo[i]; b[i + 8] = b_hi[i]; }
            acc[nt] = __builtin_amdgcn_wmma_f32_16x16x32_bf16(
                false, a, false, b, (short)0, acc[nt], false, false);
        }
    }
    // C/D layout: VGPR r holds D[M = r + 8*half][N = l]
    #pragma unroll
    for (int nt = 0; nt < 8; ++nt) {
        float bv = bias[nt * 16 + l];
        float* out = D + (size_t)m0 * 128 + nt * 16 + l;
        #pragma unroll
        for (int r = 0; r < 8; ++r)
            out[(size_t)(r + 8 * half) * 128] = acc[nt][r] + bv;
    }
}

// ---- 4. LIF scan (decay_input, soft reset): v += (x-v)*sigmoid(w); s=H(v-1); v-=s
__global__ void lif_scan_spike(const float* __restrict__ h, const float* __restrict__ wv,
                               bf16_t* __restrict__ s) {
    int idx = blockIdx.x * blockDim.x + threadIdx.x;  // over NB*NH1
    float sig = 1.0f / (1.0f + __expf(-wv[0]));
    float v = 0.0f;
    const int stride = NB * NH1;
    for (int t = 0; t < NT; ++t) {
        float xv = h[(size_t)t * stride + idx];
        v = fmaf(xv - v, sig, v);
        float sp = (v - 1.0f >= 0.0f) ? 1.0f : 0.0f;
        v -= sp;
        s[(size_t)t * stride + idx] = (bf16_t)sp;
    }
}

// ---- 5. Final LIF scan fused with mean over T -> out[b][g]
__global__ void lif_scan_mean(const float* __restrict__ h, const float* __restrict__ wv,
                              float* __restrict__ out) {
    int idx = blockIdx.x * blockDim.x + threadIdx.x;  // over NB*NH2
    float sig = 1.0f / (1.0f + __expf(-wv[0]));
    float v = 0.0f, acc = 0.0f;
    const int stride = NB * NH2;
    for (int t = 0; t < NT; ++t) {
        float xv = h[(size_t)t * stride + idx];
        v = fmaf(xv - v, sig, v);
        float sp = (v - 1.0f >= 0.0f) ? 1.0f : 0.0f;
        v -= sp;
        acc += sp;
    }
    out[idx] = acc * (1.0f / (float)NT);
}

extern "C" void kernel_launch(void* const* d_in, const int* in_sizes, int n_in,
                              void* d_out, int out_size, void* d_ws, size_t ws_size,
                              hipStream_t stream) {
    const float* x      = (const float*)d_in[0];
    const float* conv_w = (const float*)d_in[1];
    const float* conv_b = (const float*)d_in[2];
    const float* fc1_w  = (const float*)d_in[3];
    const float* fc1_b  = (const float*)d_in[4];
    const float* fc2_w  = (const float*)d_in[5];
    const float* fc2_b  = (const float*)d_in[6];
    const float* w1     = (const float*)d_in[7];
    const float* w2     = (const float*)d_in[8];
    float* out = (float*)d_out;

    char* ws = (char*)d_ws;
    bf16_t* Apk   = (bf16_t*)(ws + OFF_APK);
    float*  h12   = (float*) (ws + OFF_H12);
    bf16_t* s1    = (bf16_t*)(ws + OFF_S1);
    bf16_t* W1bf  = (bf16_t*)(ws + OFF_W1BF);
    bf16_t* fc2bf = (bf16_t*)(ws + OFF_FC2BF);
    float*  b1eff = (float*) (ws + OFF_B1EFF);

    // 1. fold conv into fc1 + bf16 weight conversion
    {
        int n = NH1 * K1P + NH1 + NH2 * NH1;
        prep_weights<<<(n + 255) / 256, 256, 0, stream>>>(
            conv_w, conv_b, fc1_w, fc1_b, fc2_w, W1bf, b1eff, fc2bf);
    }
    // 2. pack x windows -> bf16 A matrix [t*NB+b][K1P]
    {
        size_t n = (size_t)MROWS * K1P;
        pack_A<<<(unsigned)((n + 255) / 256), 256, 0, stream>>>(x, Apk);
    }
    // 3. GEMM1: h1[t,b,h] = Apk @ W1eff^T + b1eff   (M=51200, N=128, K=608)
    gemm_wmma_bf16<K1P><<<MROWS / 128, 256, 0, stream>>>(Apk, W1bf, b1eff, h12);
    // 4. LIF scan 1 -> spikes (bf16, exact 0/1)
    lif_scan_spike<<<(NB * NH1) / 256, 256, 0, stream>>>(h12, w1, s1);
    // 5. GEMM2: h2[t,b,g] = s1 @ fc2_w^T + fc2_b    (M=51200, N=128, K=128), reuse h buffer
    gemm_wmma_bf16<NH1><<<MROWS / 128, 256, 0, stream>>>(s1, fc2bf, fc2_b, h12);
    // 6. LIF scan 2 + temporal mean -> out[b][g]
    lif_scan_mean<<<(NB * NH2) / 256, 256, 0, stream>>>(h12, w2, out);
}
